// SEModule_78658031059208
// MI455X (gfx1250) — matrix operand order repaired and checked by reading, not compiled, not run on hardware
//
#include <hip/hip_runtime.h>
#include <hip/hip_bf16.h>

typedef __attribute__((ext_vector_type(16))) _Float16 v16h;
typedef __attribute__((ext_vector_type(8)))  float    v8f;
typedef __attribute__((ext_vector_type(4)))  float    v4f;   // native vector for NT builtins

#define BN_EPS 1e-5f

// ---------------------------------------------------------------------------
// Kernel 1: squeeze — per-(b,c) mean over the 64x64 plane (4096 floats).
// One 256-thread block per plane; float4 loads; LDS tree reduction.
// Regular-temporal loads on purpose: this pass warms L2 (192 MB) so the
// excite pass (which runs in REVERSE order) can hit the still-resident tail.
// ---------------------------------------------------------------------------
__global__ void squeeze_kernel(const float* __restrict__ in, float* __restrict__ s) {
    __shared__ float red[256];
    const int plane = blockIdx.x;                       // b*512 + c
    const float4* p = (const float4*)(in + (size_t)plane * 4096);
    float sum = 0.f;
#pragma unroll
    for (int i = 0; i < 4; ++i) {
        float4 v = p[threadIdx.x + i * 256];
        sum += (v.x + v.y) + (v.z + v.w);
    }
    red[threadIdx.x] = sum;
    __syncthreads();
#pragma unroll
    for (int off = 128; off > 0; off >>= 1) {
        if (threadIdx.x < (unsigned)off) red[threadIdx.x] += red[threadIdx.x + off];
        __syncthreads();
    }
    if (threadIdx.x == 0) s[plane] = red[0] * (1.0f / 4096.0f);
}

// ---------------------------------------------------------------------------
// WMMA 16x16 f32 tile:  D += A[16xK] * B[Kx16]
//   A row-major (M x K), lda given.   B[k][n] = W[n*ldw + k]  (W row-major NxK)
// Fragment assembly follows the CDNA5 16-bit layouts (CDNA5 ISA §7.12.2):
//   A: M = lane&15; element e -> K = e + (e&8) + (lane>=16 ? 8 : 0)
//   B: N = lane&15; element e -> K = e + (e&8) + (lane>=16 ? 8 : 0)
// ---------------------------------------------------------------------------
__device__ inline v8f gemm_tile_f16(const float* A, int lda,
                                    const float* W, int ldw,
                                    int K, v8f acc) {
    const int lane   = threadIdx.x & 31;
    const int m      = lane & 15;
    const int n      = lane & 15;
    const int ka_off = (lane & 16) ? 8  : 0;
    const int kb_off = (lane & 16) ? 8  : 0;
    for (int k0 = 0; k0 < K; k0 += 32) {
        v16h a, b;
#pragma unroll
        for (int e = 0; e < 16; ++e) {
            const int ka = k0 + e + (e & 8) + ka_off;
            const int kb = k0 + e + (e & 8) + kb_off;
            a[e] = (_Float16)A[m * lda + ka];
            b[e] = (_Float16)W[n * ldw + kb];
        }
        // 8 args: (neg_a, A, neg_b, B, c_mod, C, reuse_a, reuse_b)
        acc = __builtin_amdgcn_wmma_f32_16x16x32_f16(false, a, false, b,
                                                     (short)0, acc, false, false);
    }
    return acc;
}

// ---------------------------------------------------------------------------
// Kernel 2: gate — h = relu(s @ w1^T + b1); BN over batch; g = sigmoid(h @ w2^T + b2)
// Single block, 256 threads = 8 waves. M=32 (batch), Cb=128, C=512.
// GEMM1: 2x8  tiles of 16x16, K=512  -> 2 tiles/wave
// GEMM2: 2x32 tiles of 16x16, K=128  -> 8 tiles/wave
// f16 inputs / f32 accumulate; BN statistics kept in fp32 in LDS.
// ---------------------------------------------------------------------------
__global__ void gate_kernel(const float* __restrict__ s,
                            const float* __restrict__ w1, const float* __restrict__ b1,
                            const float* __restrict__ gamma, const float* __restrict__ beta,
                            const float* __restrict__ w2, const float* __restrict__ b2,
                            float* __restrict__ g) {
    __shared__ float h[32 * 128];
    const int wave = threadIdx.x >> 5;
    const int lane = threadIdx.x & 31;
    const int rbase = (lane & 16) ? 8 : 0;   // D layout: VGPR r -> M = r (+8 for hi lanes)
    const int col16 = lane & 15;             // D layout: N = lane&15

    // ---- GEMM1: h[32][128] = relu(s[32][512] * w1^T + b1) ----
    for (int t = wave * 2; t < wave * 2 + 2; ++t) {
        const int mt = t >> 3;          // 0..1
        const int nt = t & 7;           // 0..7
        v8f acc = {};
        acc = gemm_tile_f16(s + mt * 16 * 512, 512, w1 + nt * 16 * 512, 512, 512, acc);
        const int col = nt * 16 + col16;
        const float bias = b1[col];
#pragma unroll
        for (int r = 0; r < 8; ++r) {
            const int row = mt * 16 + rbase + r;
            h[row * 128 + col] = fmaxf(acc[r] + bias, 0.0f);
        }
    }
    __syncthreads();

    // ---- BatchNorm over batch dim (32 rows), per column; fp32 stats ----
    if (threadIdx.x < 128) {
        const int j = threadIdx.x;
        float sum = 0.f, sumsq = 0.f;
#pragma unroll
        for (int i = 0; i < 32; ++i) {
            const float v = h[i * 128 + j];
            sum += v; sumsq += v * v;
        }
        const float mu  = sum * (1.0f / 32.0f);
        const float var = sumsq * (1.0f / 32.0f) - mu * mu;
        const float sc  = rsqrtf(var + BN_EPS) * gamma[j];
        const float sh  = beta[j] - mu * sc;
#pragma unroll
        for (int i = 0; i < 32; ++i)
            h[i * 128 + j] = h[i * 128 + j] * sc + sh;
    }
    __syncthreads();

    // ---- GEMM2: g[32][512] = sigmoid(h[32][128] * w2^T + b2) ----
    for (int t = wave * 8; t < wave * 8 + 8; ++t) {
        const int mt = t >> 5;          // 0..1
        const int nt = t & 31;          // 0..31
        v8f acc = {};
        acc = gemm_tile_f16(h + mt * 16 * 128, 128, w2 + nt * 16 * 128, 128, 128, acc);
        const int col = nt * 16 + col16;
        const float bias = b2[col];
#pragma unroll
        for (int r = 0; r < 8; ++r) {
            const int row = mt * 16 + rbase + r;
            const float v = acc[r] + bias;
            g[row * 512 + col] = 1.0f / (1.0f + expf(-v));
        }
    }
}

// ---------------------------------------------------------------------------
// Kernel 3: excite — out = in * g[b*512+c], float4 streaming.
// 1024 float4 per plane -> plane = (float4 index) >> 10.
// L2-residency tuning for MI455X (192 MB L2 vs 268 MB tensor):
//   * REVERSED block order: the squeeze pass just streamed the input in
//     ascending order, so L2 holds the ~192 MB tail. Reading the tail first
//     converts a large fraction of this second pass into L2 hits.
//   * Non-temporal input load: last use of the input — hit if resident, but
//     don't retain the line.
//   * Non-temporal output store: the 268 MB output must not evict the warm
//     input lines we are about to read.
// ---------------------------------------------------------------------------
__global__ void excite_kernel(const float* __restrict__ in,
                              const float* __restrict__ g,
                              float* __restrict__ out,
                              unsigned nblocks) {
    // reverse block order at 4 KB (256-float4) granularity; coalesced within block
    const size_t i4 = (size_t)(nblocks - 1u - blockIdx.x) * blockDim.x + threadIdx.x;
    const int plane = (int)(i4 >> 10);
    const float gain = g[plane];
    v4f v = __builtin_nontemporal_load(&((const v4f*)in)[i4]);
    v *= gain;
    __builtin_nontemporal_store(v, &((v4f*)out)[i4]);
}

// ---------------------------------------------------------------------------
extern "C" void kernel_launch(void* const* d_in, const int* in_sizes, int n_in,
                              void* d_out, int out_size, void* d_ws, size_t ws_size,
                              hipStream_t stream) {
    const float* input = (const float*)d_in[0];   // [32,512,64,64]
    const float* w1    = (const float*)d_in[1];   // [128,512]
    const float* b1    = (const float*)d_in[2];   // [128]
    const float* gamma = (const float*)d_in[3];   // [128]
    const float* beta  = (const float*)d_in[4];   // [128]
    const float* w2    = (const float*)d_in[5];   // [512,128]
    const float* b2    = (const float*)d_in[6];   // [512]
    float* out = (float*)d_out;

    float* s = (float*)d_ws;          // [32*512] squeeze means
    float* g = s + 32 * 512;          // [32*512] gates

    const int planes = 32 * 512;                       // 16384
    squeeze_kernel<<<planes, 256, 0, stream>>>(input, s);
    gate_kernel<<<1, 256, 0, stream>>>(s, w1, b1, gamma, beta, w2, b2, g);

    const size_t n4 = (size_t)32 * 512 * 64 * 64 / 4;  // 16,777,216 float4
    const unsigned nblocks = (unsigned)(n4 / 256);     // 65,536 blocks
    excite_kernel<<<nblocks, 256, 0, stream>>>(input, g, out, nblocks);
}
